// TransformerDecoderLayer_20624432956088
// MI455X (gfx1250) — matrix operand order repaired and checked by reading, not compiled, not run on hardware
//
#include <hip/hip_runtime.h>

// ---------------------------------------------------------------------------
// CDNA5 (gfx1250) transformer decoder layer, bf16 WMMA everywhere.
// wave32; WMMA fragment layouts per CDNA5 ISA 7.12.2.
// ---------------------------------------------------------------------------

typedef __bf16 bf16;
typedef __attribute__((ext_vector_type(16))) __bf16 v16bf;
typedef __attribute__((ext_vector_type(8)))  float  v8f;

union FragU { v16bf v; uint4 q[2]; };

__device__ __forceinline__ v8f v8f_zero() {
  v8f z;
#pragma unroll
  for (int i = 0; i < 8; ++i) z[i] = 0.0f;
  return z;
}

// A fragment: 16x32 bf16 tile (MxK), row-major source with leading dim lda.
// lanes 0-15: elems 0..7 -> K=0..7, elems 8..15 -> K=16..23  (row = lane)
// lanes16-31: elems 0..7 -> K=8..15, elems 8..15 -> K=24..31
__device__ __forceinline__ v16bf load_a_frag(const bf16* A, size_t lda,
                                             int row0, int M, int k0) {
  int lane = threadIdx.x & 31;
  int m = lane & 15, h = lane >> 4;
  int r = row0 + m; r = (r < M) ? r : (M - 1);
  const bf16* p = A + (size_t)r * lda + (size_t)(k0 + h * 8);
  FragU f;
  f.q[0] = *reinterpret_cast<const uint4*>(p);
  f.q[1] = *reinterpret_cast<const uint4*>(p + 16);
  return f.v;
}

// B fragment for C[m][n] = sum_k A[m][k] * W[n][k]  (i.e. X @ W^T).
// B[k][n] = W[n][k]; lane holds column n = lane&15, K = 8*half..+7 then 16+8*half..+7.
__device__ __forceinline__ v16bf load_b_frag(const bf16* W, size_t ldw,
                                             int n0, int N, int k0) {
  int lane = threadIdx.x & 31;
  int n = n0 + (lane & 15); n = (n < N) ? n : (N - 1);
  int h = lane >> 4;
  const bf16* p = W + (size_t)n * ldw + (size_t)(k0 + h * 8);
  FragU f;
  f.q[0] = *reinterpret_cast<const uint4*>(p);
  f.q[1] = *reinterpret_cast<const uint4*>(p + 16);
  return f.v;
}

__device__ __forceinline__ v8f wmma_bf16(v16bf a, v16bf b, v8f c) {
  return __builtin_amdgcn_wmma_f32_16x16x32_bf16(false, a, false, b,
                                                 (short)0, c, false, false);
}

// ---------------------------------------------------------------------------
// Generic batched GEMM:  C = act(A @ W^T + bias)
// z = blockIdx.z decomposed as z1 = z / Z2, z2 = z % Z2; each operand gets
// offset z1*s?1 + z2*s?2.  Wave tile = (TM*16) x (TN*16); WG = MW x NW waves.
// ---------------------------------------------------------------------------
template <int MW, int NW, int TM, int TN, bool RELU, bool BF16OUT, bool TRANSC>
__global__ __launch_bounds__(MW * NW * 32) void k_gemm(
    const bf16* __restrict__ A, size_t lda, size_t sA1, size_t sA2,
    const bf16* __restrict__ W, size_t ldw, size_t sW1, size_t sW2,
    const float* __restrict__ bias,
    void* __restrict__ C, size_t ldc, size_t sC1, size_t sC2,
    int M, int N, int K, int Z2) {
  int z  = blockIdx.z;
  int z1 = z / Z2, z2 = z - z1 * Z2;
  A += (size_t)z1 * sA1 + (size_t)z2 * sA2;
  W += (size_t)z1 * sW1 + (size_t)z2 * sW2;
  size_t coff = (size_t)z1 * sC1 + (size_t)z2 * sC2;

  int wave = threadIdx.x >> 5;
  int wm = wave / NW, wn = wave - wm * NW;
  int m0 = blockIdx.y * (MW * TM * 16) + wm * (TM * 16);
  int n0 = blockIdx.x * (NW * TN * 16) + wn * (TN * 16);

  v8f acc[TM][TN];
#pragma unroll
  for (int i = 0; i < TM; ++i)
#pragma unroll
    for (int j = 0; j < TN; ++j) acc[i][j] = v8f_zero();

  for (int k0 = 0; k0 < K; k0 += 32) {
    v16bf af[TM], bfr[TN];
#pragma unroll
    for (int i = 0; i < TM; ++i) af[i] = load_a_frag(A, lda, m0 + i * 16, M, k0);
#pragma unroll
    for (int j = 0; j < TN; ++j) bfr[j] = load_b_frag(W, ldw, n0 + j * 16, N, k0);
#pragma unroll
    for (int i = 0; i < TM; ++i)
#pragma unroll
      for (int j = 0; j < TN; ++j) acc[i][j] = wmma_bf16(af[i], bfr[j], acc[i][j]);
  }

  int lane = threadIdx.x & 31;
  int nl = lane & 15, hl = lane >> 4;
  float* Cf = (float*)C;
  bf16*  Cb = (bf16*)C;
#pragma unroll
  for (int i = 0; i < TM; ++i) {
#pragma unroll
    for (int j = 0; j < TN; ++j) {
      int ncol = n0 + j * 16 + nl;
      float bv = 0.0f;
      if (bias) { int nc = (ncol < N) ? ncol : (N - 1); bv = bias[nc]; }
#pragma unroll
      for (int v = 0; v < 8; ++v) {
        int r = m0 + i * 16 + v + 8 * hl;
        if (r < M && ncol < N) {
          float x = acc[i][j][v] + bv;
          if (RELU) x = (x > 0.0f) ? x : 0.0f;
          size_t idx = TRANSC ? (coff + (size_t)ncol * ldc + (size_t)r)
                              : (coff + (size_t)r * ldc + (size_t)ncol);
          if (BF16OUT) Cb[idx] = (bf16)x; else Cf[idx] = x;
        }
      }
    }
  }
}

// ---------------------------------------------------------------------------
// Fused get_relevant_semantic: one WG (256 thr = 8 waves) per (b,t).
// H = relu(fc[b,t] + fs[b,:] + b0) built in LDS slab-by-slab (64x32 bf16),
// H2 = relu(H @ W1^T + b1) kept in accumulators (each wave owns 64 cols),
// scores = H2 @ w2 folded into epilogue (shfl-xor + LDS atomics),
// softmax over S=64, rel = a^T @ sem written bf16 in (t,b,D) order.
// ---------------------------------------------------------------------------
__global__ __launch_bounds__(256) void k_relsem(
    const float* __restrict__ fc, const float* __restrict__ fs,
    const float* __restrict__ b0, const bf16* __restrict__ w1b,
    const float* __restrict__ b1, const float* __restrict__ w2,
    const float* __restrict__ sem, bf16* __restrict__ relb) {
  const int Dd = 512, Sd = 64, Bd = 32, Td = 64;
  int bt = blockIdx.x;
  int b = bt / Td, t = bt - b * Td;

  __shared__ bf16  hs[64][32];
  __shared__ float sc[64];
  __shared__ float red;

  int tid = threadIdx.x;
  int lane = tid & 31, wave = tid >> 5;
  int nl = lane & 15, hl = lane >> 4;
  int n0 = wave * 64;

  const float* fcrow  = fc + (size_t)bt * Dd;
  const float* fsbase = fs + (size_t)b * Sd * Dd;

  if (tid < 64) sc[tid] = 0.0f;

  v8f acc[4][4];
#pragma unroll
  for (int i = 0; i < 4; ++i)
#pragma unroll
    for (int j = 0; j < 4; ++j) acc[i][j] = v8f_zero();

  int srow = tid >> 2;
  int kk   = (tid & 3) * 8;
  for (int k0 = 0; k0 < Dd; k0 += 32) {
    __syncthreads();
    const float* fsp = fsbase + (size_t)srow * Dd + k0 + kk;
#pragma unroll
    for (int e = 0; e < 8; ++e) {
      float v = fcrow[k0 + kk + e] + fsp[e] + b0[k0 + kk + e];
      hs[srow][kk + e] = (bf16)((v > 0.0f) ? v : 0.0f);
    }
    __syncthreads();
    v16bf af[4], bfr[4];
#pragma unroll
    for (int i = 0; i < 4; ++i) af[i] = load_a_frag(&hs[0][0], 32, i * 16, 64, 0);
#pragma unroll
    for (int j = 0; j < 4; ++j) bfr[j] = load_b_frag(w1b, Dd, n0 + j * 16, Dd, k0);
#pragma unroll
    for (int i = 0; i < 4; ++i)
#pragma unroll
      for (int j = 0; j < 4; ++j) acc[i][j] = wmma_bf16(af[i], bfr[j], acc[i][j]);
  }

  // epilogue: relu(acc + b1) dotted with w2, reduced to per-row scores
  float part[4][8];
#pragma unroll
  for (int i = 0; i < 4; ++i)
#pragma unroll
    for (int v = 0; v < 8; ++v) part[i][v] = 0.0f;
#pragma unroll
  for (int i = 0; i < 4; ++i)
#pragma unroll
    for (int j = 0; j < 4; ++j) {
      int ncol = n0 + j * 16 + nl;
      float bv = b1[ncol], wv = w2[ncol];
#pragma unroll
      for (int v = 0; v < 8; ++v) {
        float x = acc[i][j][v] + bv;
        x = (x > 0.0f) ? x : 0.0f;
        part[i][v] += x * wv;
      }
    }
#pragma unroll
  for (int i = 0; i < 4; ++i)
#pragma unroll
    for (int v = 0; v < 8; ++v) {
      float s = part[i][v];
      for (int m = 1; m < 16; m <<= 1) s += __shfl_xor(s, m, 32);
      if (nl == 0) atomicAdd(&sc[i * 16 + v + 8 * hl], s);
    }
  __syncthreads();

  if (tid == 0) {
    float mx = sc[0];
    for (int s2 = 1; s2 < 64; ++s2) mx = fmaxf(mx, sc[s2]);
    float sum = 0.0f;
    for (int s2 = 0; s2 < 64; ++s2) { float e = __expf(sc[s2] - mx); sc[s2] = e; sum += e; }
    red = 1.0f / sum;
  }
  __syncthreads();

  float inv = red;
  const float* semb = sem + (size_t)b * Sd * Dd;
  for (int d = tid; d < Dd; d += 256) {
    float a = 0.0f;
    for (int s2 = 0; s2 < 64; ++s2) a += sc[s2] * semb[(size_t)s2 * Dd + d];
    relb[((size_t)t * Bd + b) * Dd + d] = (bf16)(a * inv);
  }
}

// ---------------------------------------------------------------------------
// Row softmax: f32 scores -> bf16 attention, scale applied pre-exp. Nk<=256.
// ---------------------------------------------------------------------------
__global__ __launch_bounds__(256) void k_softmax(const float* __restrict__ S,
                                                 bf16* __restrict__ Aout,
                                                 int Nk, float scale) {
  int row = blockIdx.x, tid = threadIdx.x;
  const float* p = S + (size_t)row * Nk;
  __shared__ float sh[256];
  float v = (tid < Nk) ? p[tid] * scale : -1e30f;
  sh[tid] = v; __syncthreads();
  for (int s = 128; s > 0; s >>= 1) { if (tid < s) sh[tid] = fmaxf(sh[tid], sh[tid + s]); __syncthreads(); }
  float mx = sh[0]; __syncthreads();
  float e = (tid < Nk) ? __expf(v - mx) : 0.0f;
  sh[tid] = e; __syncthreads();
  for (int s = 128; s > 0; s >>= 1) { if (tid < s) sh[tid] += sh[tid + s]; __syncthreads(); }
  float inv = 1.0f / sh[0];
  if (tid < Nk) Aout[(size_t)row * Nk + tid] = (bf16)(e * inv);
}

// ---------------------------------------------------------------------------
// out = LayerNorm(a + b) * g + be ; writes f32 (and bf16 copy when outb!=0)
// ---------------------------------------------------------------------------
__global__ __launch_bounds__(256) void k_addln(const float* __restrict__ a,
                                               const float* __restrict__ b,
                                               const float* __restrict__ g,
                                               const float* __restrict__ be,
                                               float* __restrict__ outf,
                                               bf16* __restrict__ outb) {
  int row = blockIdx.x, tid = threadIdx.x;
  const float* pa = a + (size_t)row * 512;
  const float* pb = b + (size_t)row * 512;
  float x0 = pa[tid] + pb[tid];
  float x1 = pa[tid + 256] + pb[tid + 256];
  __shared__ float sh[256];
  sh[tid] = x0 + x1; __syncthreads();
  for (int s = 128; s > 0; s >>= 1) { if (tid < s) sh[tid] += sh[tid + s]; __syncthreads(); }
  float mu = sh[0] * (1.0f / 512.0f);
  __syncthreads();
  sh[tid] = (x0 - mu) * (x0 - mu) + (x1 - mu) * (x1 - mu);
  __syncthreads();
  for (int s = 128; s > 0; s >>= 1) { if (tid < s) sh[tid] += sh[tid + s]; __syncthreads(); }
  float rstd = rsqrtf(sh[0] * (1.0f / 512.0f) + 1e-5f);
  float y0 = (x0 - mu) * rstd * g[tid] + be[tid];
  float y1 = (x1 - mu) * rstd * g[tid + 256] + be[tid + 256];
  size_t base = (size_t)row * 512;
  outf[base + tid] = y0; outf[base + tid + 256] = y1;
  if (outb) { outb[base + tid] = (bf16)y0; outb[base + tid + 256] = (bf16)y1; }
}

__global__ void k_add_dual(const float* __restrict__ a, const float* __restrict__ b,
                           float* __restrict__ outf, bf16* __restrict__ outb, int n) {
  int i = blockIdx.x * blockDim.x + threadIdx.x;
  int stride = gridDim.x * blockDim.x;
  for (; i < n; i += stride) { float v = a[i] + b[i]; outf[i] = v; outb[i] = (bf16)v; }
}

__global__ void k_cvt(const float* __restrict__ src, bf16* __restrict__ dst, int n) {
  int i = blockIdx.x * blockDim.x + threadIdx.x;
  int stride = gridDim.x * blockDim.x;
  for (; i < n; i += stride) dst[i] = (bf16)src[i];
}

// ---------------------------------------------------------------------------
// Host orchestration
// ---------------------------------------------------------------------------
namespace {

struct Bump {
  char* p; size_t off;
  template <typename T> T* get(size_t n) {
    T* r = (T*)(p + off);
    off = (off + n * sizeof(T) + 255) & ~(size_t)255;
    return r;
  }
};

template <int MW, int NW, int TM, int TN, bool RELU, bool BF16OUT, bool TRANSC>
void launch_gemm(hipStream_t st,
                 const bf16* A, size_t lda, size_t sA1, size_t sA2,
                 const bf16* W, size_t ldw, size_t sW1, size_t sW2,
                 const float* bias, void* C, size_t ldc, size_t sC1, size_t sC2,
                 int M, int N, int K, int Z, int Z2) {
  dim3 grid((N + NW * TN * 16 - 1) / (NW * TN * 16),
            (M + MW * TM * 16 - 1) / (MW * TM * 16), Z);
  dim3 block(MW * NW * 32, 1, 1);
  k_gemm<MW, NW, TM, TN, RELU, BF16OUT, TRANSC><<<grid, block, 0, st>>>(
      A, lda, sA1, sA2, W, ldw, sW1, sW2, bias, C, ldc, sC1, sC2, M, N, K, Z2);
}

void cvt(hipStream_t st, const float* s, bf16* d, int n) {
  int blocks = (n + 2047) / 2048; if (blocks > 8192) blocks = 8192;
  k_cvt<<<blocks, 256, 0, st>>>(s, d, n);
}

// one multi-head-attention block: q_in (2048x512 bf16, rows (t*B+b)),
// kv (Skv*B x 512 bf16, rows (s*B+b)), Skv in {64, 256}
void run_mha(hipStream_t st, const bf16* q_in, const bf16* kv, int Skv,
             const bf16* wqkv, const float* bqkv, const bf16* wo, const float* bo,
             bf16* qb, bf16* kb, bf16* vtb, float* scores, bf16* attb,
             bf16* ob, float* proj) {
  const int B = 32, H = 8, D = 512, Tq = 64;
  const size_t BD = (size_t)B * D;
  // Q/K projections (bf16 out)
  launch_gemm<2, 4, 4, 2, false, true, false>(st, q_in, D, 0, 0, wqkv, D, 0, 0,
      bqkv, qb, D, 0, 0, 2048, D, D, 1, 1);
  launch_gemm<2, 4, 4, 2, false, true, false>(st, kv, D, 0, 0, wqkv + 512 * 512, D, 0, 0,
      bqkv + 512, kb, D, 0, 0, Skv * B, D, D, 1, 1);
  // V projection, transposed per batch: vtb[b][d][s]  (batched z=b)
  launch_gemm<2, 4, 4, 2, false, true, true>(st, kv, BD, 0, D,
      wqkv + 2 * 512 * 512, D, 0, 0, bqkv + 1024,
      vtb, (size_t)Skv, 0, (size_t)D * Skv, Skv, D, D, B, B);
  // scores = Q @ K^T per (b,h): z1=b, z2=h
  launch_gemm<2, 2, 2, 2, false, false, false>(st,
      qb, BD, D, 64, kb, BD, D, 64, nullptr,
      scores, (size_t)Skv, (size_t)H * Tq * Skv, (size_t)Tq * Skv,
      Tq, Skv, 64, B * H, H);
  // softmax rows (scale = 1/sqrt(64))
  k_softmax<<<B * H * Tq, 256, 0, st>>>(scores, attb, Skv, 0.125f);
  // out = att @ Vt^T per (b,h) -> ob rows (q*B+b), cols h*64+..
  launch_gemm<2, 2, 2, 2, false, true, false>(st,
      attb, (size_t)Skv, (size_t)H * Tq * Skv, (size_t)Tq * Skv,
      vtb, (size_t)Skv, (size_t)D * Skv, (size_t)64 * Skv, nullptr,
      ob, BD, D, 64, Tq, 64, Skv, B * H, H);
  // output projection (f32 out)
  launch_gemm<2, 4, 4, 2, false, false, false>(st, ob, D, 0, 0, wo, D, 0, 0,
      bo, proj, D, 0, 0, 2048, D, D, 1, 1);
}

} // namespace

extern "C" void kernel_launch(void* const* d_in, const int* in_sizes, int n_in,
                              void* d_out, int out_size, void* d_ws, size_t ws_size,
                              hipStream_t stream) {
  (void)in_sizes; (void)n_in; (void)out_size; (void)ws_size;
  const int D = 512, B = 32, T = 64, S = 64, M = 256, FF = 2048, H = 8;
  const int NT = T * B;            // 2048 tokens
  const float* tgt  = (const float*)d_in[0];
  const float* mem  = (const float*)d_in[1];
  const float* sem  = (const float*)d_in[2];
  const float* wqkv_f[3] = {(const float*)d_in[3],  (const float*)d_in[7],  (const float*)d_in[11]};
  const float* bqkv[3]   = {(const float*)d_in[4],  (const float*)d_in[8],  (const float*)d_in[12]};
  const float* wo_f[3]   = {(const float*)d_in[5],  (const float*)d_in[9],  (const float*)d_in[13]};
  const float* bo[3]     = {(const float*)d_in[6],  (const float*)d_in[10], (const float*)d_in[14]};
  const float* mlp_w0 = (const float*)d_in[15];
  const float* mlp_b0 = (const float*)d_in[16];
  const float* mlp_w1 = (const float*)d_in[17];
  const float* mlp_b1 = (const float*)d_in[18];
  const float* mlp_w2 = (const float*)d_in[19];
  const float* ffn_w1 = (const float*)d_in[21];
  const float* ffn_b1 = (const float*)d_in[22];
  const float* ffn_w2 = (const float*)d_in[23];
  const float* ffn_b2 = (const float*)d_in[24];
  const float* ln_g[3] = {(const float*)d_in[25], (const float*)d_in[27], (const float*)d_in[29]};
  const float* ln_b[3] = {(const float*)d_in[26], (const float*)d_in[28], (const float*)d_in[30]};
  float* out = (float*)d_out;

  Bump ws{(char*)d_ws, 0};
  bf16* tgtb = ws.get<bf16>((size_t)NT * D);
  bf16* memb = ws.get<bf16>((size_t)M * B * D);
  bf16* semb = ws.get<bf16>((size_t)B * S * D);
  bf16* wqkvb[3]; for (int i = 0; i < 3; ++i) wqkvb[i] = ws.get<bf16>((size_t)3 * D * D);
  bf16* wob[3];   for (int i = 0; i < 3; ++i) wob[i]   = ws.get<bf16>((size_t)D * D);
  bf16* wm0b = ws.get<bf16>((size_t)D * 2 * D);
  bf16* wm1b = ws.get<bf16>((size_t)D * D);
  bf16* wf1b = ws.get<bf16>((size_t)FF * D);
  bf16* wf2b = ws.get<bf16>((size_t)D * FF);
  float* fc = ws.get<float>((size_t)NT * D);
  float* fs = ws.get<float>((size_t)B * S * D);
  bf16* relb = ws.get<bf16>((size_t)NT * D);
  bf16* qb   = ws.get<bf16>((size_t)NT * D);
  bf16* kb   = ws.get<bf16>((size_t)M * B * D);
  bf16* vtb  = ws.get<bf16>((size_t)M * B * D);
  float* scores = ws.get<float>((size_t)B * H * T * M);
  bf16*  attb   = ws.get<bf16>((size_t)B * H * T * M);
  bf16*  ob     = ws.get<bf16>((size_t)NT * D);
  float* proj   = ws.get<float>((size_t)NT * D);
  float* xaf = ws.get<float>((size_t)NT * D);
  float* xbf = ws.get<float>((size_t)NT * D);
  bf16*  xab = ws.get<bf16>((size_t)NT * D);
  bf16*  xbb = ws.get<bf16>((size_t)NT * D);
  bf16*  f1b = ws.get<bf16>((size_t)NT * FF);

  // ---- bf16 conversions -------------------------------------------------
  cvt(stream, tgt, tgtb, NT * D);
  cvt(stream, mem, memb, M * B * D);
  cvt(stream, sem, semb, B * S * D);
  for (int i = 0; i < 3; ++i) { cvt(stream, wqkv_f[i], wqkvb[i], 3 * D * D);
                                cvt(stream, wo_f[i],   wob[i],   D * D); }
  cvt(stream, mlp_w0, wm0b, D * 2 * D);
  cvt(stream, mlp_w1, wm1b, D * D);
  cvt(stream, ffn_w1, wf1b, FF * D);
  cvt(stream, ffn_w2, wf2b, D * FF);

  // ---- get_relevant_semantic -------------------------------------------
  // fc[b*T+t] = feats(b,t) @ w0c^T : batched over b (feats = tgt transposed)
  launch_gemm<2, 4, 4, 2, false, false, false>(stream,
      tgtb, (size_t)B * D, 0, (size_t)D, wm0b, 2 * D, 0, 0, nullptr,
      fc, D, 0, (size_t)T * D, T, D, D, B, B);
  // fs[b*S+s] = sem @ w0s^T (w0s = w0[:, D:])
  launch_gemm<2, 4, 4, 2, false, false, false>(stream,
      semb, D, 0, 0, wm0b + D, 2 * D, 0, 0, nullptr,
      fs, D, 0, 0, B * S, D, D, 1, 1);
  k_relsem<<<NT, 256, 0, stream>>>(fc, fs, mlp_b0, wm1b, mlp_b1, mlp_w2, sem, relb);

  // ---- pt: cross-attn on relevant semantics; x0 = tgt + mha(...) --------
  run_mha(stream, tgtb, relb, S, wqkvb[0], bqkv[0], wob[0], bo[0],
          qb, kb, vtb, scores, attb, ob, proj);
  k_add_dual<<<2048, 256, 0, stream>>>(tgt, proj, xaf, xab, NT * D);

  // ---- self-attention; x1 = LN1(x0 + mha(x0,x0,x0)) ---------------------
  run_mha(stream, xab, xab, T, wqkvb[1], bqkv[1], wob[1], bo[1],
          qb, kb, vtb, scores, attb, ob, proj);
  k_addln<<<NT, 256, 0, stream>>>(xaf, proj, ln_g[0], ln_b[0], xbf, xbb);

  // ---- cross-attention on memory; x2 = LN2(x1 + mha(x1,mem,mem)) --------
  run_mha(stream, xbb, memb, M, wqkvb[2], bqkv[2], wob[2], bo[2],
          qb, kb, vtb, scores, attb, ob, proj);
  k_addln<<<NT, 256, 0, stream>>>(xbf, proj, ln_g[1], ln_b[1], xaf, xab);

  // ---- FFN; out = LN3(x2 + ffn(x2)) -------------------------------------
  launch_gemm<2, 4, 4, 2, true, true, false>(stream, xab, D, 0, 0,
      wf1b, D, 0, 0, ffn_b1, f1b, FF, 0, 0, NT, FF, D, 1, 1);
  launch_gemm<2, 4, 4, 2, false, false, false>(stream, f1b, FF, 0, 0,
      wf2b, FF, 0, 0, ffn_b2, proj, D, 0, 0, NT, D, FF, 1, 1);
  k_addln<<<NT, 256, 0, stream>>>(xaf, proj, ln_g[2], ln_b[2], out, nullptr);
}
